// ESupConLoss_17918603559474
// MI455X (gfx1250) — matrix operand order repaired and checked by reading, not compiled, not run on hardware
//
#include <hip/hip_runtime.h>
#include <hip/hip_bf16.h>

// ESupConLoss on MI455X (gfx1250).
//
// Three 8192x8192x128 similarity matmuls streamed through
// V_WMMA_F32_16X16X32_BF16 with a precomputed bf16 hi/lo split
// (a*b ~= ah*bh + ah*bl + al*bh, ~2^-16 relative error), fused with an
// online (running-max, tile-max amortized) logsumexp in the log2 domain
// (v_exp_f32 is base-2; log2(e) folded in via one v_fma per logit).
// The hi/lo split is computed ONCE by a prep kernel into a K-permuted
// layout so WMMA operand fragments are single contiguous 32B loads.
// All reductions are fixed-order (no atomics) => deterministic.

#define NROW   8192
#define DDIM   128
#define NSLICE 4              // gridDim.y split of the j-sweep
#define JT_TOTAL (NROW / 16)  // 512 j-tiles of 16
#define ALPHA_C 2.0f

constexpr float LOG2E = 1.44269504088896340736f;
constexpr float LN2   = 0.69314718055994530942f;
#define MASKVAL (-1.0e30f)

typedef __bf16 v16bf __attribute__((ext_vector_type(16)));
typedef __bf16 v8bf  __attribute__((ext_vector_type(8)));
typedef float  v8f   __attribute__((ext_vector_type(8)));

__device__ __forceinline__ v8f wmma_bf16(v16bf a, v16bf b, v8f c) {
  // D = A(16x32) * B(32x16) + C, f32 accumulate
  return __builtin_amdgcn_wmma_f32_16x16x32_bf16(
      /*neg_a=*/false, a, /*neg_b=*/false, b,
      /*c_mod=*/(short)0, c, /*reuse_a=*/false, /*reuse_b=*/false);
}

__device__ __forceinline__ v16bf cat8(v8bf lo, v8bf hi) {
  v16bf r;
#pragma unroll
  for (int e = 0; e < 8; ++e) { r[e] = lo[e]; r[8 + e] = hi[e]; }
  return r;
}

// ---------------------------------------------------------------------------
// Prep: f32 -> (bf16 hi, bf16 lo) split, stored in fragment-permuted K order.
// Per row, chunk ck = ks*2 + c (16 elems):
//   pos 0..7  <- K = ks*32 + c*8 + e
//   pos 8..15 <- K = ks*32 + 16 + c*8 + e
// Then an A fragment (ks, lane-half hl) == contiguous chunk (ks,hl) @ 32B,
// and a B fragment is two contiguous 16B runs.
// ---------------------------------------------------------------------------
__global__ __launch_bounds__(256) void prep_split(
    const float* __restrict__ zau, const float* __restrict__ ztp,
    __bf16* __restrict__ au_h, __bf16* __restrict__ au_l,
    __bf16* __restrict__ tp_h, __bf16* __restrict__ tp_l) {
  const int t   = blockIdx.x * 256 + threadIdx.x;  // 2 * 8192 * 8 threads
  const int mat = t >> 16;                         // 0 = au, 1 = tp
  const int rem = t & 0xFFFF;
  const int row = rem >> 3;
  const int ck  = rem & 7;
  const int ks  = ck >> 1;
  const int c   = ck & 1;
  const float* src = (mat ? ztp : zau) + (size_t)row * DDIM;
  __bf16* dh = (mat ? tp_h : au_h) + (size_t)row * DDIM + ck * 16;
  __bf16* dl = (mat ? tp_l : au_l) + (size_t)row * DDIM + ck * 16;
  const int kA = ks * 32 + c * 8;
  const int kB = kA + 16;
  v8bf h0, l0, h1, l1;
#pragma unroll
  for (int e = 0; e < 8; ++e) {
    float x   = src[kA + e];
    __bf16 hb = (__bf16)x;
    h0[e] = hb; l0[e] = (__bf16)(x - (float)hb);
    x  = src[kB + e];
    hb = (__bf16)x;
    h1[e] = hb; l1[e] = (__bf16)(x - (float)hb);
  }
  *(v8bf*)(dh)     = h0;
  *(v8bf*)(dh + 8) = h1;
  *(v8bf*)(dl)     = l0;
  *(v8bf*)(dl + 8) = l1;
}

// Online-logsumexp tile update. Logits arrive in natural domain; state (m,s)
// lives in log2 domain: exp2(v*log2e - m) is a single fma + v_exp_f32.
template <bool DIAG>
__device__ __forceinline__ void tile_update(
    const v8f& caa, const v8f& ctt, const v8f& cat_, int i_col, int j0, int hl,
    float& m_pos, float& s_pos, float& m_neg, float& s_neg) {
  float va[8], vt[8], vx[8];
#pragma unroll
  for (int p = 0; p < 8; ++p) {
    const bool dg = DIAG && ((j0 + p + 8 * hl) == i_col);
    va[p] = dg ? MASKVAL : caa[p];
    vt[p] = dg ? MASKVAL : ctt[p];
    vx[p] = dg ? MASKVAL : cat_[p];
  }
  float tm = MASKVAL, tn = MASKVAL;
#pragma unroll
  for (int p = 0; p < 8; ++p) {
    tm = fmaxf(tm, fmaxf(va[p], vt[p]));
    tn = fmaxf(tn, vx[p]);
  }
  const float nm = fmaxf(m_pos, tm * LOG2E);
  const float nn = fmaxf(m_neg, tn * LOG2E);
  float addp = 0.f, addn = 0.f;
#pragma unroll
  for (int p = 0; p < 8; ++p) {
    addp += exp2f(fmaf(va[p], LOG2E, -nm)) + exp2f(fmaf(vt[p], LOG2E, -nm));
    addn += exp2f(fmaf(vx[p], LOG2E, -nn));
  }
  s_pos = fmaf(s_pos, exp2f(m_pos - nm), addp);
  s_neg = fmaf(s_neg, exp2f(m_neg - nn), addn);
  m_pos = nm;
  m_neg = nn;
}

// ---------------------------------------------------------------------------
// Main: fused WMMA matmul + online logsumexp. Wave owns 16 i-columns as the
// stationary B operand; sweeps j-tiles of its gridDim.y slice as A.
// C element (VGPR p, lane ln): row j0+p+8*(ln>=16), col i0+(ln&15).
// ---------------------------------------------------------------------------
__global__ __launch_bounds__(256, 1) void supcon_main(
    const __bf16* __restrict__ au_h, const __bf16* __restrict__ au_l,
    const __bf16* __restrict__ tp_h, const __bf16* __restrict__ tp_l,
    float* __restrict__ mp_ws, float* __restrict__ sp_ws,
    float* __restrict__ mn_ws, float* __restrict__ sn_ws) {
  const int ln  = threadIdx.x & 31;
  const int wid = threadIdx.x >> 5;
  const int hl  = ln >> 4;
  const int l15 = ln & 15;
  const int i0  = blockIdx.x * 128 + wid * 16;
  const int jt0 = blockIdx.y * (JT_TOTAL / NSLICE);
  const int jt1 = jt0 + (JT_TOTAL / NSLICE);

  // Stationary i-side fragments (B operand layout), loop-invariant.
  v16bf b_au_h[4], b_au_l[4], b_tp_h[4], b_tp_l[4];
  {
    const __bf16* rah = au_h + (size_t)(i0 + l15) * DDIM;
    const __bf16* ral = au_l + (size_t)(i0 + l15) * DDIM;
    const __bf16* rth = tp_h + (size_t)(i0 + l15) * DDIM;
    const __bf16* rtl = tp_l + (size_t)(i0 + l15) * DDIM;
#pragma unroll
    for (int ks = 0; ks < 4; ++ks) {
      const int o0 = ks * 32 + hl * 16;
      const int o1 = o0 + 8;
      b_au_h[ks] = cat8(*(const v8bf*)(rah + o0), *(const v8bf*)(rah + o1));
      b_au_l[ks] = cat8(*(const v8bf*)(ral + o0), *(const v8bf*)(ral + o1));
      b_tp_h[ks] = cat8(*(const v8bf*)(rth + o0), *(const v8bf*)(rth + o1));
      b_tp_l[ks] = cat8(*(const v8bf*)(rtl + o0), *(const v8bf*)(rtl + o1));
    }
  }

  float m_pos = MASKVAL, s_pos = 0.f;
  float m_neg = MASKVAL, s_neg = 0.f;
  const int i_col = i0 + l15;

#pragma unroll 1
  for (int jt = jt0; jt < jt1; ++jt) {
    const int j0 = jt * 16;
    const size_t jrow = (size_t)(j0 + l15) * DDIM + (size_t)(hl * 16);
    const __bf16* jah = au_h + jrow;
    const __bf16* jal = au_l + jrow;
    const __bf16* jth = tp_h + jrow;
    const __bf16* jtl = tp_l + jrow;
    if (jt + 1 < jt1) {
      // WGP-scope (locality 3) prefetch: warm L0 for next tile's rows.
      __builtin_prefetch(jah + (size_t)16 * DDIM, 0, 3);
      __builtin_prefetch(jal + (size_t)16 * DDIM, 0, 3);
      __builtin_prefetch(jth + (size_t)16 * DDIM, 0, 3);
      __builtin_prefetch(jtl + (size_t)16 * DDIM, 0, 3);
    }

    v8f caa = {}, ctt = {}, cat_ = {};
#pragma unroll
    for (int ks = 0; ks < 4; ++ks) {
      // A fragment = one contiguous 32B chunk in the permuted layout.
      const v16bf a_au_h = *(const v16bf*)(jah + ks * 32);
      const v16bf a_au_l = *(const v16bf*)(jal + ks * 32);
      const v16bf a_tp_h = *(const v16bf*)(jth + ks * 32);
      const v16bf a_tp_l = *(const v16bf*)(jtl + ks * 32);

      // s_aa(i,j): A=au_j, B=au_i   (3-term bf16 split)
      caa = wmma_bf16(a_au_h, b_au_h[ks], caa);
      caa = wmma_bf16(a_au_h, b_au_l[ks], caa);
      caa = wmma_bf16(a_au_l, b_au_h[ks], caa);
      // s_tt(i,j): A=tp_j, B=tp_i
      ctt = wmma_bf16(a_tp_h, b_tp_h[ks], ctt);
      ctt = wmma_bf16(a_tp_h, b_tp_l[ks], ctt);
      ctt = wmma_bf16(a_tp_l, b_tp_h[ks], ctt);
      // s_at(i,j) = au_i . tp_j: A=tp_j, B=au_i
      cat_ = wmma_bf16(a_tp_h, b_au_h[ks], cat_);
      cat_ = wmma_bf16(a_tp_h, b_au_l[ks], cat_);
      cat_ = wmma_bf16(a_tp_l, b_au_h[ks], cat_);
    }

    // Diagonal only possible when j-tile == i-tile (wave-uniform branch).
    if (j0 == i0)
      tile_update<true>(caa, ctt, cat_, i_col, j0, hl, m_pos, s_pos, m_neg, s_neg);
    else
      tile_update<false>(caa, ctt, cat_, i_col, j0, hl, m_pos, s_pos, m_neg, s_neg);
  }

  // Lanes l and l+16 hold the same i-column over disjoint j rows: combine.
  {
    float om = __shfl_xor(m_pos, 16);
    float os = __shfl_xor(s_pos, 16);
    float nm = fmaxf(m_pos, om);
    s_pos = s_pos * exp2f(m_pos - nm) + os * exp2f(om - nm);
    m_pos = nm;
    om = __shfl_xor(m_neg, 16);
    os = __shfl_xor(s_neg, 16);
    nm = fmaxf(m_neg, om);
    s_neg = s_neg * exp2f(m_neg - nm) + os * exp2f(om - nm);
    m_neg = nm;
  }
  if (hl == 0) {
    const size_t idx = (size_t)blockIdx.y * NROW + (size_t)i_col;
    mp_ws[idx] = m_pos;
    sp_ws[idx] = s_pos;
    mn_ws[idx] = m_neg;
    sn_ws[idx] = s_neg;
  }
}

// Merge the NSLICE partial (m,s) pairs per row; emit neg_i - pos_i (nats).
__global__ __launch_bounds__(256) void supcon_merge(
    const float* __restrict__ mp, const float* __restrict__ sp,
    const float* __restrict__ mn, const float* __restrict__ sn,
    float* __restrict__ rowA) {
  const int i = blockIdx.x * blockDim.x + threadIdx.x;
  float m1 = MASKVAL, m2 = MASKVAL;
#pragma unroll
  for (int s = 0; s < NSLICE; ++s) {
    m1 = fmaxf(m1, mp[(size_t)s * NROW + i]);
    m2 = fmaxf(m2, mn[(size_t)s * NROW + i]);
  }
  float a1 = 0.f, a2 = 0.f;
#pragma unroll
  for (int s = 0; s < NSLICE; ++s) {
    a1 += sp[(size_t)s * NROW + i] * exp2f(mp[(size_t)s * NROW + i] - m1);
    a2 += sn[(size_t)s * NROW + i] * exp2f(mn[(size_t)s * NROW + i] - m2);
  }
  const float pos = m1 + log2f(a1);
  const float neg = m2 + log2f(a2);
  rowA[i] = LN2 * (neg - pos);
}

// Per-row diagonal dot + prototype term:
//   ALPHA * (au_i . tp_i) + ((au_i - tp_i) . (pt1 - pt0)) / N
__global__ __launch_bounds__(256) void diag_pt(
    const float* __restrict__ zau, const float* __restrict__ ztp,
    const float* __restrict__ fc, float* __restrict__ rowB) {
  const int ln  = threadIdx.x & 31;
  const int wid = threadIdx.x >> 5;
  const int i   = blockIdx.x * 8 + wid;
  const float4 a  = ((const float4*)(zau + (size_t)i * DDIM))[ln];
  const float4 t  = ((const float4*)(ztp + (size_t)i * DDIM))[ln];
  const float4 p0 = ((const float4*)fc)[ln];
  const float4 p1 = ((const float4*)(fc + DDIM))[ln];
  float dot = a.x * t.x + a.y * t.y + a.z * t.z + a.w * t.w;
  float prj = (a.x - t.x) * (p1.x - p0.x) + (a.y - t.y) * (p1.y - p0.y) +
              (a.z - t.z) * (p1.z - p0.z) + (a.w - t.w) * (p1.w - p0.w);
#pragma unroll
  for (int off = 16; off > 0; off >>= 1) {
    dot += __shfl_xor(dot, off);
    prj += __shfl_xor(prj, off);
  }
  if (ln == 0) rowB[i] = ALPHA_C * dot + prj * (1.0f / (float)NROW);
}

// Deterministic single-block final reduction -> scalar loss.
__global__ __launch_bounds__(256) void final_reduce(
    const float* __restrict__ rowA, const float* __restrict__ rowB,
    float* __restrict__ out) {
  __shared__ float sm[256];
  float acc = 0.f;
  for (int i = threadIdx.x; i < NROW; i += 256) acc += rowA[i] + rowB[i];
  sm[threadIdx.x] = acc;
  __syncthreads();
#pragma unroll
  for (int s = 128; s > 0; s >>= 1) {
    if ((int)threadIdx.x < s) sm[threadIdx.x] += sm[threadIdx.x + s];
    __syncthreads();
  }
  if (threadIdx.x == 0) out[0] = sm[0] / (float)(NROW + 2);
}

extern "C" void kernel_launch(void* const* d_in, const int* in_sizes, int n_in,
                              void* d_out, int out_size, void* d_ws, size_t ws_size,
                              hipStream_t stream) {
  (void)in_sizes; (void)n_in; (void)out_size; (void)ws_size;
  const float* zau = (const float*)d_in[0];
  const float* ztp = (const float*)d_in[1];
  const float* fc  = (const float*)d_in[2];
  // d_in[3] = labels (int64) — unused by the reference math.

  // Workspace layout: 4 bf16 matrices (8 MB) + f32 state (~0.6 MB).
  __bf16* au_h = (__bf16*)d_ws;
  __bf16* au_l = au_h + (size_t)NROW * DDIM;
  __bf16* tp_h = au_l + (size_t)NROW * DDIM;
  __bf16* tp_l = tp_h + (size_t)NROW * DDIM;
  float*  mp   = (float*)(tp_l + (size_t)NROW * DDIM);
  float*  sp   = mp + (size_t)NSLICE * NROW;
  float*  mn   = sp + (size_t)NSLICE * NROW;
  float*  sn   = mn + (size_t)NSLICE * NROW;
  float*  rowA = sn + (size_t)NSLICE * NROW;
  float*  rowB = rowA + NROW;

  prep_split<<<(2 * NROW * 8) / 256, 256, 0, stream>>>(zau, ztp, au_h, au_l, tp_h, tp_l);
  supcon_main<<<dim3(NROW / 128, NSLICE), 256, 0, stream>>>(au_h, au_l, tp_h, tp_l,
                                                            mp, sp, mn, sn);
  supcon_merge<<<NROW / 256, 256, 0, stream>>>(mp, sp, mn, sn, rowA);
  diag_pt<<<NROW / 8, 256, 0, stream>>>(zau, ztp, fc, rowB);
  final_reduce<<<1, 256, 0, stream>>>(rowA, rowB, (float*)d_out);
}
